// AdmittanceEncoder_36704790511703
// MI455X (gfx1250) — compile-verified, matrix operand order checked by reading, not run on hardware
//
#include <hip/hip_runtime.h>

typedef __attribute__((ext_vector_type(16))) _Float16 v16h;
typedef __attribute__((ext_vector_type(2)))  _Float16 h2;
typedef __attribute__((ext_vector_type(8)))  float    v8f;

constexpr int NN = 100000;   // nodes   (divisible by 16; 6250 tiles)
constexpr int EE = 200000;   // edges   (divisible by 16; 12500 tiles)
constexpr int BG = 1024;     // graphs

// ---- WMMA fragment index helpers (CDNA5 16-bit A layout, wave32) ----
// A 16x32 f16: lanes 0-15 -> M=lane, lanes 16-31 -> M=lane-16 (other K half).
// pair p (2 f16 per VGPR): K base = ((p>>2)<<4) | (half<<3) | ((p&3)<<1)
__device__ __forceinline__ int a_k0(int p, int half) {
  return ((p >> 2) << 4) | (half << 3) | ((p & 3) << 1);
}

__device__ __forceinline__ v8f wmma16(v16h a, v16h b, v8f c) {
  return __builtin_amdgcn_wmma_f32_16x16x32_f16(false, a, false, b,
                                                (short)0, c, false, false);
}

// Intra-wave LDS RAW fence: DS ops are in-order within a wave; we only need
// to stop the compiler from reordering and (belt-and-braces) drain DScnt.
__device__ __forceinline__ void lds_sync() {
  asm volatile("" ::: "memory");
#if __has_builtin(__builtin_amdgcn_s_wait_dscnt)
  __builtin_amdgcn_s_wait_dscnt(0);
#else
  __syncthreads();
#endif
  asm volatile("" ::: "memory");
}

// ---------------------------------------------------------------------------
// Pack a 64x64 row-major f32 weight matrix into per-lane B fragments:
// out[((kb*4+nt)*32+lane)*8+p] packs f16{W[k,n], W[k+1,n]} with
// n = nt*16 + (lane&15), k = kb*32 + a_k0(p, lane>>4).
// ---------------------------------------------------------------------------
__global__ void pack_b_kernel(const float* __restrict__ W,
                              unsigned* __restrict__ out) {
  int i = blockIdx.x * blockDim.x + threadIdx.x;   // grid covers exactly 2048
  int p    = i & 7;
  int lane = (i >> 3) & 31;
  int nt   = (i >> 8) & 3;
  int kb   = i >> 10;
  int half = lane >> 4;
  int n = nt * 16 + (lane & 15);
  int k = kb * 32 + a_k0(p, half);
  _Float16 lo = (_Float16)W[k * 64 + n];
  _Float16 hi = (_Float16)W[(k + 1) * 64 + n];
  unsigned u = (unsigned)__builtin_bit_cast(unsigned short, lo) |
               ((unsigned)__builtin_bit_cast(unsigned short, hi) << 16);
  out[i] = u;
}

// node types = argmax of one-hot x  --------------------------------------
__global__ void types_kernel(const float* __restrict__ x, int* __restrict__ nt, int n_) {
  int n = blockIdx.x * blockDim.x + threadIdx.x;
  if (n >= n_) return;
  float best = x[n * 4];
  int t = 0;
#pragma unroll
  for (int j = 1; j < 4; ++j) {
    float v = x[n * 4 + j];
    if (v > best) { best = v; t = j; }
  }
  nt[n] = t;
}

// layer-0 lin: hn = onehot(x) @ lin0  -> row lookup -----------------------
__global__ void hn0_kernel(const int* __restrict__ ntype,
                           const float* __restrict__ lin0,
                           _Float16* __restrict__ hn16) {
  long i = (long)blockIdx.x * blockDim.x + threadIdx.x;   // exactly N*64
  int n = (int)(i >> 6), c = (int)(i & 63);
  hn16[i] = (_Float16)lin0[ntype[n] * 64 + c];
}

// agg init with bias ------------------------------------------------------
__global__ void init_agg_kernel(float* __restrict__ agg,
                                const float* __restrict__ bias) {
  long i = (long)blockIdx.x * blockDim.x + threadIdx.x;   // exactly N*64
  agg[i] = bias[i & 63];
}

__global__ void zero_kernel(float* __restrict__ p) {
  long i = (long)blockIdx.x * blockDim.x + threadIdx.x;   // exact
  p[i] = 0.f;
}

// ---------------------------------------------------------------------------
// hn = h16 @ lin (64x64) via WMMA; 2 waves/block, one 16-row tile per wave.
// ---------------------------------------------------------------------------
__global__ __launch_bounds__(64)
void lin_rows_kernel(const _Float16* __restrict__ h16,
                     const unsigned* __restrict__ linpk,
                     _Float16* __restrict__ hn16) {
  const int lane = threadIdx.x & 31;
  const int tile = blockIdx.x * 2 + (threadIdx.x >> 5);
  const int half = lane >> 4, nlane = lane & 15;
  const _Float16* arow = h16 + (long)(tile * 16 + nlane) * 64;
  v16h a0, a1;
#pragma unroll
  for (int p = 0; p < 8; ++p) {
    int k0 = a_k0(p, half);
    a0[2 * p] = arow[k0];      a0[2 * p + 1] = arow[k0 + 1];
    a1[2 * p] = arow[32 + k0]; a1[2 * p + 1] = arow[32 + k0 + 1];
  }
#pragma unroll
  for (int nt = 0; nt < 4; ++nt) {
    v8f c;
#pragma unroll
    for (int i = 0; i < 8; ++i) c[i] = 0.f;
    v16h b0 = *(const v16h*)(linpk + ((0 * 4 + nt) * 32 + lane) * 8);
    v16h b1 = *(const v16h*)(linpk + ((1 * 4 + nt) * 32 + lane) * 8);
    c = wmma16(a0, b0, c);
    c = wmma16(a1, b1, c);
#pragma unroll
    for (int i = 0; i < 8; ++i) {
      int gr = tile * 16 + i + half * 8;
      hn16[(long)gr * 64 + nt * 16 + nlane] = (_Float16)c[i];
    }
  }
}

// ---------------------------------------------------------------------------
// Edge message kernel: 4 waves/block, one 16-edge tile per wave.
//   xj = hn16[src]  (16x64 f16 A-fragments, gathered)
//   for phi in {G,C,L}: t = relu(xj@W1+b1); msg += (bc_phi * t) @ W2
//   atomicAdd msg into agg[dst]
// Intermediate re-layout (C-frag -> A-frag) goes through a per-wave LDS slice.
// ---------------------------------------------------------------------------
__global__ __launch_bounds__(128)
void edge_msg_kernel(const _Float16* __restrict__ hn16,
                     const int* __restrict__ src, const int* __restrict__ dst,
                     const float* __restrict__ ea,
                     const unsigned* __restrict__ bpkL,   // 6*2048 u32
                     const float* __restrict__ b1G, const float* __restrict__ b1C,
                     const float* __restrict__ b1L,
                     float* __restrict__ agg) {
  __shared__ _Float16 tmp[4][16][64];
  __shared__ int   srcs[4][16];
  __shared__ int   dsts[4][16];
  __shared__ float bcs[4][16][3];

  const int lane = threadIdx.x & 31;
  const int w    = threadIdx.x >> 5;
  const int tile = blockIdx.x * 4 + w;
  const int half = lane >> 4, nlane = lane & 15;

  // warm next block's edge metadata while we compute (global_prefetch_b8)
  __builtin_prefetch(src + (tile + 4) * 16, 0, 0);
  __builtin_prefetch(dst + (tile + 4) * 16, 0, 0);
  __builtin_prefetch(ea + (long)(tile + 4) * 48, 0, 0);

  if (lane < 16) {
    int e = tile * 16 + lane;
    srcs[w][lane] = src[e];
    dsts[w][lane] = dst[e];
#pragma unroll
    for (int j = 0; j < 3; ++j)
      bcs[w][lane][j] = 2.f * (sqrtf(1.f + ea[(long)e * 3 + j]) - 1.f);
  }
  lds_sync();

  // gather A = xj
  const _Float16* arow = hn16 + (long)srcs[w][nlane] * 64;
  v16h aj0, aj1;
#pragma unroll
  for (int p = 0; p < 8; ++p) {
    int k0 = a_k0(p, half);
    aj0[2 * p] = arow[k0];      aj0[2 * p + 1] = arow[k0 + 1];
    aj1[2 * p] = arow[32 + k0]; aj1[2 * p + 1] = arow[32 + k0 + 1];
  }

  v8f macc[4];
#pragma unroll
  for (int nt = 0; nt < 4; ++nt)
#pragma unroll
    for (int i = 0; i < 8; ++i) macc[nt][i] = 0.f;

  const float* b1s[3] = {b1G, b1C, b1L};
#pragma unroll
  for (int phi = 0; phi < 3; ++phi) {
    const unsigned* W1p = bpkL + (phi * 2) * 2048;
    const unsigned* W2p = bpkL + (phi * 2 + 1) * 2048;

    // t = relu(xj @ W1 + b1) -> LDS (f16, row-major 16x64)
#pragma unroll
    for (int nt = 0; nt < 4; ++nt) {
      float bv = b1s[phi][nt * 16 + nlane];
      v8f c;
#pragma unroll
      for (int i = 0; i < 8; ++i) c[i] = bv;
      v16h b0 = *(const v16h*)(W1p + ((0 * 4 + nt) * 32 + lane) * 8);
      v16h b1 = *(const v16h*)(W1p + ((1 * 4 + nt) * 32 + lane) * 8);
      c = wmma16(aj0, b0, c);
      c = wmma16(aj1, b1, c);
#pragma unroll
      for (int i = 0; i < 8; ++i)
        tmp[w][i + half * 8][nt * 16 + nlane] = (_Float16)fmaxf(c[i], 0.f);
    }
    lds_sync();

    // reload as A-fragment, scaled by per-edge Box-Cox coefficient (pk f16)
    _Float16 bch = (_Float16)bcs[w][nlane][phi];
    h2 bc2 = {bch, bch};
    const h2* row2 = (const h2*)tmp[w][nlane];
    v16h a0, a1;
#pragma unroll
    for (int p = 0; p < 8; ++p) {
      int k0 = a_k0(p, half);
      h2 v0 = row2[k0 >> 1] * bc2;
      h2 v1 = row2[(32 + k0) >> 1] * bc2;
      a0[2 * p] = v0[0]; a0[2 * p + 1] = v0[1];
      a1[2 * p] = v1[0]; a1[2 * p + 1] = v1[1];
    }
#pragma unroll
    for (int nt = 0; nt < 4; ++nt) {
      v16h b0 = *(const v16h*)(W2p + ((0 * 4 + nt) * 32 + lane) * 8);
      v16h b1 = *(const v16h*)(W2p + ((1 * 4 + nt) * 32 + lane) * 8);
      macc[nt] = wmma16(a0, b0, macc[nt]);
      macc[nt] = wmma16(a1, b1, macc[nt]);
    }
    lds_sync();
  }

  // segment-sum scatter (all edges valid: E % 16 == 0)
#pragma unroll
  for (int nt = 0; nt < 4; ++nt) {
#pragma unroll
    for (int i = 0; i < 8; ++i) {
      int d = dsts[w][i + half * 8];
      atomicAdd(&agg[(long)d * 64 + nt * 16 + nlane], macc[nt][i]);
    }
  }
}

// ---------------------------------------------------------------------------
// LayerNorm + ReLU + residual; one wave per node, 2 cols/lane. N % 8 == 0.
// ---------------------------------------------------------------------------
__global__ __launch_bounds__(256)
void ln_res_kernel(const float* __restrict__ agg,
                   const float* __restrict__ lng, const float* __restrict__ lnb,
                   float* __restrict__ h32, _Float16* __restrict__ h16o,
                   const int* __restrict__ ntype,
                   const float* __restrict__ projW, const float* __restrict__ projb,
                   int is_layer0) {
  int lane = threadIdx.x & 31;
  int n = blockIdx.x * 8 + (threadIdx.x >> 5);
  int c0 = lane, c1 = lane + 32;
  float v0 = agg[(long)n * 64 + c0];
  float v1 = agg[(long)n * 64 + c1];
  float s = v0 + v1;
#pragma unroll
  for (int off = 16; off >= 1; off >>= 1) s += __shfl_xor(s, off, 32);
  float mean = s * (1.f / 64.f);
  float d0 = v0 - mean, d1 = v1 - mean;
  float q = d0 * d0 + d1 * d1;
#pragma unroll
  for (int off = 16; off >= 1; off >>= 1) q += __shfl_xor(q, off, 32);
  float rstd = rsqrtf(q * (1.f / 64.f) + 1e-5f);
  float h0 = d0 * rstd * lng[c0] + lnb[c0];
  float h1 = d1 * rstd * lng[c1] + lnb[c1];
  float r0, r1;
  if (is_layer0) {
    int t = ntype[n];
    r0 = projW[t * 64 + c0] + projb[c0];
    r1 = projW[t * 64 + c1] + projb[c1];
  } else {
    r0 = h32[(long)n * 64 + c0];
    r1 = h32[(long)n * 64 + c1];
  }
  float o0 = fmaxf(h0, 0.f) + r0;
  float o1 = fmaxf(h1, 0.f) + r1;
  h32[(long)n * 64 + c0] = o0;
  h32[(long)n * 64 + c1] = o1;
  h16o[(long)n * 64 + c0] = (_Float16)o0;
  h16o[(long)n * 64 + c1] = (_Float16)o1;
}

// terminal readout scatter (last-writer-wins on duplicates, like .at[].set)
__global__ __launch_bounds__(256)
void scatter_terms_kernel(const float* __restrict__ h32,
                          const int* __restrict__ ntype,
                          const int* __restrict__ batch,
                          float* __restrict__ vin, float* __restrict__ vout,
                          float* __restrict__ gnd) {
  int lane = threadIdx.x & 31;
  int n = blockIdx.x * 8 + (threadIdx.x >> 5);
  int t = ntype[n];
  float* base;
  if (t == 1) base = vin;
  else if (t == 2) base = vout;
  else if (t == 0) base = gnd;
  else return;
  int g = batch[n];
  base[(long)g * 64 + lane]      = h32[(long)n * 64 + lane];
  base[(long)g * 64 + lane + 32] = h32[(long)n * 64 + lane + 32];
}

// heads: z|mu|logvar  out[3][B][5]; one thread per graph
__global__ __launch_bounds__(256)
void heads_kernel(const float* __restrict__ vin, const float* __restrict__ vout,
                  const float* __restrict__ gnd, const float* __restrict__ P,
                  float* __restrict__ out) {
  int g = blockIdx.x * blockDim.x + threadIdx.x;   // exactly BG threads
  const float* vi = vin  + (long)g * 64;
  const float* vo = vout + (long)g * 64;
  const float* gn = gnd  + (long)g * 64;
  // param offsets within flat P (jax pytree order)
  const int mu_topo_W = 646, mu_topo_b = 1030;
  const int mu_vin_W = 1032, mu_vin_b = 1096;
  const int mu_vout_W = 1097, mu_vout_b = 1161;
  const int mu_gnd_W = 581, mu_gnd_b = 645;
  const int lv_topo_W = 65, lv_topo_b = 449;
  const int lv_vin_W = 451, lv_vin_b = 515;
  const int lv_vout_W = 516, lv_vout_b = 580;
  const int lv_gnd_W = 0, lv_gnd_b = 64;

  float mu[5], lv[5];
  mu[0] = P[mu_topo_b];     mu[1] = P[mu_topo_b + 1];
  lv[0] = P[lv_topo_b];     lv[1] = P[lv_topo_b + 1];
  mu[2] = P[mu_vin_b];      lv[2] = P[lv_vin_b];
  mu[3] = P[mu_vout_b];     lv[3] = P[lv_vout_b];
  mu[4] = P[mu_gnd_b];      lv[4] = P[lv_gnd_b];
  for (int i = 0; i < 64; ++i) {
    float a = vi[i], b = vo[i], c = gn[i];
    mu[0] += a * P[mu_topo_W + i * 2]       + b * P[mu_topo_W + (64 + i) * 2]
           + c * P[mu_topo_W + (128 + i) * 2];
    mu[1] += a * P[mu_topo_W + i * 2 + 1]   + b * P[mu_topo_W + (64 + i) * 2 + 1]
           + c * P[mu_topo_W + (128 + i) * 2 + 1];
    lv[0] += a * P[lv_topo_W + i * 2]       + b * P[lv_topo_W + (64 + i) * 2]
           + c * P[lv_topo_W + (128 + i) * 2];
    lv[1] += a * P[lv_topo_W + i * 2 + 1]   + b * P[lv_topo_W + (64 + i) * 2 + 1]
           + c * P[lv_topo_W + (128 + i) * 2 + 1];
    mu[2] += a * P[mu_vin_W + i];   lv[2] += a * P[lv_vin_W + i];
    mu[3] += b * P[mu_vout_W + i];  lv[3] += b * P[lv_vout_W + i];
    mu[4] += c * P[mu_gnd_W + i];   lv[4] += c * P[lv_gnd_W + i];
  }
#pragma unroll
  for (int j = 0; j < 5; ++j) {
    out[(long)g * 5 + j]                 = mu[j];  // z == mu at eval
    out[(long)BG * 5 + (long)g * 5 + j]  = mu[j];
    out[(long)BG * 10 + (long)g * 5 + j] = lv[j];
  }
}

// ---------------------------------------------------------------------------
extern "C" void kernel_launch(void* const* d_in, const int* in_sizes, int n_in,
                              void* d_out, int out_size, void* d_ws, size_t ws_size,
                              hipStream_t stream) {
  (void)in_sizes; (void)n_in; (void)out_size; (void)ws_size;
  const float* x   = (const float*)d_in[0];   // [N,4]
  const float* ea  = (const float*)d_in[1];   // [E,3]
  const float* P   = (const float*)d_in[2];   // flat params (84810 f32)
  const int* ei    = (const int*)d_in[3];     // [2,E]
  const int* batch = (const int*)d_in[4];     // [N]
  const int* src = ei;
  const int* dst = ei + EE;
  float* out = (float*)d_out;

  // workspace layout (bytes)
  char* ws = (char*)d_ws;
  _Float16* h16  = (_Float16*)(ws + 0);                 // N*64 f16
  _Float16* hn16 = (_Float16*)(ws + 12800000);          // N*64 f16
  float*    h32  = (float*)   (ws + 25600000);          // N*64 f32
  float*    agg  = (float*)   (ws + 51200000);          // N*64 f32
  unsigned* bpk  = (unsigned*)(ws + 76800000);          // 20*2048 u32
  float*    vin  = (float*)   (ws + 76963840);          // B*64
  float*    vout = vin + (long)BG * 64;
  float*    gnd  = vout + (long)BG * 64;
  int*      ntype= (int*)     (ws + 77750272);          // N i32

  // ---- flat param offsets (pytree: heads, layers[0..2], proj0) ----
  const int Lbase[3]  = {1162, 26378, 55434};
  const int din64[3]  = {256, 4096, 4096};
  int lin_off[3], bias_off[3], lnb_off[3], lng_off[3];
  int W1_off[3][3], W2_off[3][3], b1_off[3][3];          // [layer][phi: G,C,L]
  for (int L = 0; L < 3; ++L) {
    bias_off[L] = Lbase[L];
    lin_off[L]  = Lbase[L] + 64;
    lnb_off[L]  = Lbase[L] + 64 + din64[L];
    lng_off[L]  = Lbase[L] + 128 + din64[L];
    int phiC = Lbase[L] + 192 + din64[L];
    int base[3] = {phiC + 8256, phiC, phiC + 16512};     // G, C, L in memory
    for (int p = 0; p < 3; ++p) {
      W1_off[L][p] = base[p];
      W2_off[L][p] = base[p] + 4096;
      b1_off[L][p] = base[p] + 8192;
    }
  }
  const int proj0_W = 84490, proj0_b = 84746;

  // ---- pack all 64x64 weight matrices into WMMA B fragments ----
  // slots: L*6 + {G_W1,G_W2,C_W1,C_W2,L_W1,L_W2}; 18 = lin(layer1), 19 = lin(layer2)
  for (int L = 0; L < 3; ++L)
    for (int p = 0; p < 3; ++p) {
      pack_b_kernel<<<8, 256, 0, stream>>>(P + W1_off[L][p], bpk + (L * 6 + p * 2) * 2048);
      pack_b_kernel<<<8, 256, 0, stream>>>(P + W2_off[L][p], bpk + (L * 6 + p * 2 + 1) * 2048);
    }
  pack_b_kernel<<<8, 256, 0, stream>>>(P + lin_off[1], bpk + 18 * 2048);
  pack_b_kernel<<<8, 256, 0, stream>>>(P + lin_off[2], bpk + 19 * 2048);

  types_kernel<<<(NN + 255) / 256, 256, 0, stream>>>(x, ntype, NN);
  // layer 0: hn = onehot @ lin0 (row lookup)
  hn0_kernel<<<NN * 64 / 256, 256, 0, stream>>>(ntype, P + lin_off[0], hn16);

  for (int L = 0; L < 3; ++L) {
    if (L > 0)
      lin_rows_kernel<<<NN / 32, 64, 0, stream>>>(h16, bpk + (17 + L) * 2048, hn16);
    init_agg_kernel<<<NN * 64 / 256, 256, 0, stream>>>(agg, P + bias_off[L]);
    edge_msg_kernel<<<EE / 64, 128, 0, stream>>>(
        hn16, src, dst, ea, bpk + L * 6 * 2048,
        P + b1_off[L][0], P + b1_off[L][1], P + b1_off[L][2], agg);
    ln_res_kernel<<<NN / 8, 256, 0, stream>>>(
        agg, P + lng_off[L], P + lnb_off[L], h32, h16,
        ntype, P + proj0_W, P + proj0_b, (L == 0) ? 1 : 0);
  }

  zero_kernel<<<3 * BG * 64 / 256, 256, 0, stream>>>(vin);
  scatter_terms_kernel<<<NN / 8, 256, 0, stream>>>(h32, ntype, batch, vin, vout, gnd);
  heads_kernel<<<BG / 256, 256, 0, stream>>>(vin, vout, gnd, P, out);
}
